// LongGeminiTransformerBlock_33792802685372
// MI455X (gfx1250) — compile-verified, matrix operand order checked by reading, not run on hardware
//
#include <hip/hip_runtime.h>
#include <stdint.h>

// ---------------------------------------------------------------------------
// LongGemini transformer block on MI455X (gfx1250, wave32, WMMA bf16, TDM)
// B=1, S=2048, D=2048, H=24, Dh=128, HD=3072, F=8192
// ---------------------------------------------------------------------------

#define S_SEQ 2048
#define D_MODEL 2048
#define N_HEADS 24
#define D_HEAD 128
#define HD_DIM 3072        // N_HEADS * D_HEAD
#define F_FF 8192

typedef __bf16 v16bf __attribute__((ext_vector_type(16)));
typedef __bf16 v8bf  __attribute__((ext_vector_type(8)));
typedef float  v8f   __attribute__((ext_vector_type(8)));
typedef unsigned int u32x4 __attribute__((ext_vector_type(4)));
typedef int i32x4 __attribute__((ext_vector_type(4)));
typedef int i32x8 __attribute__((ext_vector_type(8)));

#if defined(__has_builtin)
#if __has_builtin(__builtin_amdgcn_tensor_load_to_lds) && __has_builtin(__builtin_amdgcn_s_wait_tensorcnt)
#define HAVE_TDM 1
#endif
#endif

// -------------------------- small device helpers ---------------------------

__device__ inline v8f zero8() {
    v8f z;
#pragma unroll
    for (int i = 0; i < 8; ++i) z[i] = 0.0f;
    return z;
}

// Build a v16bf A/B fragment from two contiguous 8-element (16B) runs:
// elements [0..7] <- p[0..7], elements [8..15] <- p[16..23].
// Matches CDNA5 16-bit A/B operand layout: K(e) = (e&7) + (e>=8?16:0) + laneHi*8
// when p already includes the laneHi*8 offset.
__device__ inline v16bf frag16(const __bf16* p) {
    v8bf lo = *(const v8bf*)(p);
    v8bf hi = *(const v8bf*)(p + 16);
    v16bf r;
#pragma unroll
    for (int i = 0; i < 8; ++i) { r[i] = lo[i]; r[i + 8] = hi[i]; }
    return r;
}

// Max/sum reductions across the 16-lane half-groups (lanes 0-15 / 16-31)
// using ds_swizzle group-of-32 XOR patterns (and_mask=0x1f).
__device__ inline float red_max16(float v) {
    v = fmaxf(v, __int_as_float(__builtin_amdgcn_ds_swizzle(__float_as_int(v), 0x041f)));
    v = fmaxf(v, __int_as_float(__builtin_amdgcn_ds_swizzle(__float_as_int(v), 0x081f)));
    v = fmaxf(v, __int_as_float(__builtin_amdgcn_ds_swizzle(__float_as_int(v), 0x101f)));
    v = fmaxf(v, __int_as_float(__builtin_amdgcn_ds_swizzle(__float_as_int(v), 0x201f)));
    return v;
}
__device__ inline float red_sum16(float v) {
    v += __int_as_float(__builtin_amdgcn_ds_swizzle(__float_as_int(v), 0x041f));
    v += __int_as_float(__builtin_amdgcn_ds_swizzle(__float_as_int(v), 0x081f));
    v += __int_as_float(__builtin_amdgcn_ds_swizzle(__float_as_int(v), 0x101f));
    v += __int_as_float(__builtin_amdgcn_ds_swizzle(__float_as_int(v), 0x201f));
    return v;
}

__device__ inline float gelu_tanh(float x) {
    float x3 = x * x * x;
    return 0.5f * x * (1.0f + tanhf(0.7978845608028654f * (x + 0.044715f * x3)));
}

#ifdef HAVE_TDM
// Issue a TDM 2D tile load: tile (tile_dim1=128 rows x tile_dim0=64 elems) of a
// bf16 tensor (tensor_d1 rows x tensor_d0 cols, row stride = stride0 elems)
// into LDS at byte offset lds_off.  pad_enable inserts 4 dwords (16B) after
// every 128B row -> LDS row stride 144B = 72 bf16 elems (== LDT below).
__device__ inline void tdm_load_tile_128x64(unsigned lds_off, const __bf16* gptr,
                                            unsigned tensor_d0, unsigned tensor_d1,
                                            unsigned stride0) {
    unsigned long long ga = (unsigned long long)(uintptr_t)gptr;
    u32x4 g0;
    g0[0] = 1u;                                             // count=1, user desc
    g0[1] = lds_off;                                        // lds_addr
    g0[2] = (unsigned)ga;                                   // global_addr[31:0]
    g0[3] = (unsigned)((ga >> 32) & 0x01FFFFFFull) | (2u << 30);  // addr[56:32], type=2
    i32x8 g1;
    g1[0] = (int)((1u << 16)     // data_size = 2 bytes
                | (1u << 20)     // pad_enable
                | (4u << 22)     // pad_interval: 32 dwords (128B)
                | (3u << 25));   // pad_amount: 4 dwords (16B)
    g1[1] = (int)((tensor_d0 & 0xFFFFu) << 16);             // tensor_dim0[15:0]
    g1[2] = (int)(((tensor_d0 >> 16) & 0xFFFFu) | ((tensor_d1 & 0xFFFFu) << 16));
    g1[3] = (int)(((tensor_d1 >> 16) & 0xFFFFu) | (64u << 16));   // tile_dim0 = 64
    g1[4] = 128;                                            // tile_dim1=128, tile_dim2=0
    g1[5] = (int)stride0;                                   // tensor_dim0_stride[31:0]
    g1[6] = 0;
    g1[7] = 0;
    i32x4 g2 = {0, 0, 0, 0};
    i32x4 g3 = {0, 0, 0, 0};
#if __clang_major__ >= 23
    i32x8 g4 = {0, 0, 0, 0, 0, 0, 0, 0};
    __builtin_amdgcn_tensor_load_to_lds(g0, g1, g2, g3, g4, 0);
#else
    __builtin_amdgcn_tensor_load_to_lds(g0, g1, g2, g3, 0);
#endif
}
#endif  // HAVE_TDM

// ------------------------------ LN + RMSNorm -------------------------------

__device__ inline float block_sum(float v, float* buf) {
    int t = threadIdx.x;
    buf[t] = v;
    __syncthreads();
#pragma unroll
    for (int s = 128; s > 0; s >>= 1) {
        if (t < s) buf[t] += buf[t + s];
        __syncthreads();
    }
    float r = buf[0];
    __syncthreads();
    return r;
}

__global__ __launch_bounds__(256) void ln_rms_kernel(
    const float* __restrict__ x, const float* __restrict__ g,
    const float* __restrict__ b, const float* __restrict__ ag,
    float* __restrict__ xln, __bf16* __restrict__ h) {
    __shared__ float buf[256];
    const int row = blockIdx.x;
    const float* xr = x + (size_t)row * D_MODEL;

    float s = 0.f, ss = 0.f;
#pragma unroll
    for (int j = 0; j < D_MODEL / 256; ++j) {
        float v = xr[threadIdx.x + j * 256];
        s += v;
        ss += v * v;
    }
    float S1 = block_sum(s, buf);
    float S2 = block_sum(ss, buf);
    float mu = S1 * (1.0f / D_MODEL);
    float var = S2 * (1.0f / D_MODEL) - mu * mu;
    float rstd = rsqrtf(var + 1e-5f);

    float lnv[D_MODEL / 256];
    float s2 = 0.f;
#pragma unroll
    for (int j = 0; j < D_MODEL / 256; ++j) {
        int i = threadIdx.x + j * 256;
        float v = (xr[i] - mu) * rstd * g[i] + b[i];
        lnv[j] = v;
        s2 += v * v;
        xln[(size_t)row * D_MODEL + i] = v;
    }
    float MS = block_sum(s2, buf) * (1.0f / D_MODEL);
    float r2 = rsqrtf(MS + 1e-5f);
#pragma unroll
    for (int j = 0; j < D_MODEL / 256; ++j) {
        int i = threadIdx.x + j * 256;
        h[(size_t)row * D_MODEL + i] = (__bf16)(lnv[j] * r2 * ag[i]);
    }
}

// ------------------- f32 -> bf16 transpose (weight prep) -------------------
// src[rows][cols] f32  ->  dst[cols][rows] bf16, 32x32 tiles via LDS.

__global__ __launch_bounds__(256) void cvt_transpose_bf16(
    const float* __restrict__ src, __bf16* __restrict__ dst, int rows, int cols) {
    __shared__ float tile[32][33];
    const int r0 = blockIdx.y * 32, c0 = blockIdx.x * 32;
    const int tx = threadIdx.x & 31, ty = threadIdx.x >> 5;  // 32 x 8
#pragma unroll
    for (int i = 0; i < 32; i += 8)
        tile[ty + i][tx] = src[(size_t)(r0 + ty + i) * cols + c0 + tx];
    __syncthreads();
#pragma unroll
    for (int i = 0; i < 32; i += 8)
        dst[(size_t)(c0 + ty + i) * rows + r0 + tx] = (__bf16)tile[tx][ty + i];
}

// ------------------------------- WMMA GEMM ---------------------------------
// C[M,N] = epi(A[M,K] @ Bt[N,K]^T [+bias] [gelu] [+res]).
// Block tile 128x128x64, 256 threads = 8 waves (4x2), wave tile 32x64.
// Both A and Bt tiles are staged row-contiguous (k-major) -> identical TDM /
// b128 staging, identical 2x16B fragment reads.

template <bool HAS_BIAS, bool DO_GELU, bool HAS_RES, bool OUT_F32, bool OUT_BF16,
          bool OUT_BF16T>
__global__ __launch_bounds__(256, 2) void gemm_bf16(
    const __bf16* __restrict__ A, const __bf16* __restrict__ Bt,
    const float* __restrict__ bias, const float* __restrict__ res,
    float* __restrict__ Cf, __bf16* __restrict__ Cb, __bf16* __restrict__ Cbt,
    int M, int N, int K) {
    constexpr int BM = 128, BN = 128, BK = 64, LDT = BK + 8;  // 144B LDS row
    __shared__ __bf16 As[BM * LDT];
    __shared__ __bf16 Bs[BN * LDT];

    const int tid = threadIdx.x;
    const int wave = tid >> 5, lane = tid & 31;
    const int laneLo = lane & 15, laneHi = lane >> 4;
    const int wm = wave & 3, wn = wave >> 2;
    const int rowBase = blockIdx.y * BM;
    const int colBase = blockIdx.x * BN;

    v8f acc[2][4];
#pragma unroll
    for (int i = 0; i < 2; ++i)
#pragma unroll
        for (int j = 0; j < 4; ++j) acc[i][j] = zero8();

    for (int k0 = 0; k0 < K; k0 += BK) {
        __syncthreads();
#ifdef HAVE_TDM
        if (wave == 0) {
            tdm_load_tile_128x64((unsigned)(uintptr_t)&As[0],
                                 A + (size_t)rowBase * K + k0, (unsigned)K,
                                 (unsigned)M, (unsigned)K);
            tdm_load_tile_128x64((unsigned)(uintptr_t)&Bs[0],
                                 Bt + (size_t)colBase * K + k0, (unsigned)K,
                                 (unsigned)N, (unsigned)K);
            __builtin_amdgcn_s_wait_tensorcnt(0);
        }
#else
#pragma unroll
        for (int i = 0; i < 4; ++i) {
            int c = tid + i * 256;
            int r = c >> 3, kc = (c & 7) * 8;
            *(v8bf*)(&As[r * LDT + kc]) =
                *(const v8bf*)(A + (size_t)(rowBase + r) * K + k0 + kc);
        }
#pragma unroll
        for (int i = 0; i < 4; ++i) {
            int c = tid + i * 256;
            int r = c >> 3, kc = (c & 7) * 8;
            *(v8bf*)(&Bs[r * LDT + kc]) =
                *(const v8bf*)(Bt + (size_t)(colBase + r) * K + k0 + kc);
        }
#endif
        if (k0 + BK < K) {  // prefetch next K-tile (global_prefetch_b8)
            __builtin_prefetch(A + (size_t)(rowBase + (tid >> 1)) * K + k0 + BK + (tid & 1) * 32, 0, 1);
            __builtin_prefetch(Bt + (size_t)(colBase + (tid >> 1)) * K + k0 + BK + (tid & 1) * 32, 0, 1);
        }
        __syncthreads();

#pragma unroll
        for (int kk = 0; kk < BK; kk += 32) {
            v16bf af[2], bf[4];
#pragma unroll
            for (int i = 0; i < 2; ++i)
                af[i] = frag16(&As[(wm * 32 + i * 16 + laneLo) * LDT + kk + laneHi * 8]);
#pragma unroll
            for (int j = 0; j < 4; ++j)
                bf[j] = frag16(&Bs[(wn * 64 + j * 16 + laneLo) * LDT + kk + laneHi * 8]);
#pragma unroll
            for (int i = 0; i < 2; ++i)
#pragma unroll
                for (int j = 0; j < 4; ++j)
                    acc[i][j] = __builtin_amdgcn_wmma_f32_16x16x32_bf16(
                        false, af[i], false, bf[j], (short)0, acc[i][j], false, false);
        }
    }

    // epilogue
#pragma unroll
    for (int i = 0; i < 2; ++i) {
#pragma unroll
        for (int j = 0; j < 4; ++j) {
            int mB = rowBase + wm * 32 + i * 16 + laneHi * 8;
            int n = colBase + wn * 64 + j * 16 + laneLo;
            float bv = HAS_BIAS ? bias[n] : 0.0f;
#pragma unroll
            for (int r = 0; r < 8; ++r) {
                int m = mB + r;
                float v = acc[i][j][r];
                if (HAS_BIAS) v += bv;
                if (DO_GELU) v = gelu_tanh(v);
                if (HAS_RES) v += res[(size_t)m * N + n];
                if (OUT_F32) Cf[(size_t)m * N + n] = v;
                if (OUT_BF16) Cb[(size_t)m * N + n] = (__bf16)v;
                if (OUT_BF16T) Cbt[(size_t)n * M + m] = (__bf16)v;  // [N][M]
            }
        }
    }
}

// --------------------------- flash attention (WMMA) ------------------------
// grid = (S/64, H), block = 128 (4 waves). Each wave owns a 16-query tile,
// streams 32-key tiles with online softmax. Q/K read row-major [S][HD];
// V is read pre-transposed [HD][S] so P@V B-fragments are contiguous loads.

__global__ __launch_bounds__(128) void flash_attn_kernel(
    const __bf16* __restrict__ Q, const __bf16* __restrict__ K,
    const __bf16* __restrict__ VT, __bf16* __restrict__ O) {
    const int head = blockIdx.y;
    const int wave = threadIdx.x >> 5;
    const int lane = threadIdx.x & 31;
    const int laneLo = lane & 15, laneHi = lane >> 4;
    const int qbase = blockIdx.x * 64 + wave * 16;
    const float scale = 0.08838834764831845f;  // Dh^-0.5

    __shared__ __bf16 PldsAll[4][16 * 32];
    __bf16* Plds = PldsAll[wave];

    v16bf qf[4];
    {
        const __bf16* qrow = Q + (size_t)(qbase + laneLo) * HD_DIM + head * D_HEAD + laneHi * 8;
#pragma unroll
        for (int kkI = 0; kkI < 4; ++kkI) qf[kkI] = frag16(qrow + kkI * 32);
    }

    v8f acc[8];
#pragma unroll
    for (int c = 0; c < 8; ++c) acc[c] = zero8();
    float ms[8], ls[8];
#pragma unroll
    for (int r = 0; r < 8; ++r) { ms[r] = -1e30f; ls[r] = 0.0f; }

    const int nkt = qbase / 32 + 1;  // causal
    for (int kt = 0; kt < nkt; ++kt) {
        const int kb = kt * 32;

        // scores: S(16x32) = Q(16x128) @ K^T, two 16x16 subtiles
        v8f s0 = zero8(), s1 = zero8();
#pragma unroll
        for (int kkI = 0; kkI < 4; ++kkI) {
            const __bf16* kr0 =
                K + (size_t)(kb + laneLo) * HD_DIM + head * D_HEAD + kkI * 32 + laneHi * 8;
            const __bf16* kr1 =
                K + (size_t)(kb + 16 + laneLo) * HD_DIM + head * D_HEAD + kkI * 32 + laneHi * 8;
            s0 = __builtin_amdgcn_wmma_f32_16x16x32_bf16(false, qf[kkI], false, frag16(kr0),
                                                         (short)0, s0, false, false);
            s1 = __builtin_amdgcn_wmma_f32_16x16x32_bf16(false, qf[kkI], false, frag16(kr1),
                                                         (short)0, s1, false, false);
        }

        // online softmax (C-layout row r -> global row qbase + r + laneHi*8)
#pragma unroll
        for (int r = 0; r < 8; ++r) {
            int m = qbase + r + laneHi * 8;
            float a0 = s0[r] * scale;
            float a1 = s1[r] * scale;
            if (kb + laneLo > m) a0 = -1e30f;
            if (kb + 16 + laneLo > m) a1 = -1e30f;
            float rowm = red_max16(fmaxf(a0, a1));
            float mnew = fmaxf(ms[r], rowm);
            float p0 = __expf(a0 - mnew);
            float p1 = __expf(a1 - mnew);
            float rs = red_sum16(p0 + p1);
            float alpha = __expf(ms[r] - mnew);
            ls[r] = ls[r] * alpha + rs;
            ms[r] = mnew;
#pragma unroll
            for (int c = 0; c < 8; ++c) acc[c][r] *= alpha;
            Plds[(r + laneHi * 8) * 32 + laneLo] = (__bf16)p0;
            Plds[(r + laneHi * 8) * 32 + laneLo + 16] = (__bf16)p1;
        }

        // O += P(16x32) @ V(32x128): P via LDS C->A relayout, V from VT global
        v16bf pf = frag16(Plds + laneLo * 32 + laneHi * 8);
#pragma unroll
        for (int c = 0; c < 8; ++c) {
            const __bf16* vrow =
                VT + (size_t)(head * D_HEAD + c * 16 + laneLo) * S_SEQ + kb + laneHi * 8;
            acc[c] = __builtin_amdgcn_wmma_f32_16x16x32_bf16(false, pf, false, frag16(vrow),
                                                             (short)0, acc[c], false, false);
        }
    }

    float inv[8];
#pragma unroll
    for (int r = 0; r < 8; ++r) inv[r] = 1.0f / ls[r];
#pragma unroll
    for (int c = 0; c < 8; ++c)
#pragma unroll
        for (int r = 0; r < 8; ++r)
            O[(size_t)(qbase + r + laneHi * 8) * HD_DIM + head * D_HEAD + c * 16 + laneLo] =
                (__bf16)(acc[c][r] * inv[r]);
}

// ------------------------------- launcher ----------------------------------

extern "C" void kernel_launch(void* const* d_in, const int* in_sizes, int n_in,
                              void* d_out, int out_size, void* d_ws, size_t ws_size,
                              hipStream_t stream) {
    (void)in_sizes; (void)n_in; (void)out_size; (void)ws_size;
    const float* x      = (const float*)d_in[0];
    const float* ln_g   = (const float*)d_in[1];
    const float* ln_b   = (const float*)d_in[2];
    const float* attn_g = (const float*)d_in[3];
    const float* Wq = (const float*)d_in[4];
    const float* Wk = (const float*)d_in[5];
    const float* Wv = (const float*)d_in[6];
    const float* Wo = (const float*)d_in[7];
    const float* W1 = (const float*)d_in[8];
    const float* b1 = (const float*)d_in[9];
    const float* W2 = (const float*)d_in[10];
    const float* b2 = (const float*)d_in[11];
    float* out = (float*)d_out;

    char* w = (char*)d_ws;
    auto alloc = [&](size_t bytes) -> void* {
        void* p = (void*)w;
        w += (bytes + 255) & ~(size_t)255;
        return p;
    };
    float*  xln  = (float*)alloc((size_t)S_SEQ * D_MODEL * 4);
    __bf16* h    = (__bf16*)alloc((size_t)S_SEQ * D_MODEL * 2);
    __bf16* qb   = (__bf16*)alloc((size_t)S_SEQ * HD_DIM * 2);
    __bf16* kbuf = (__bf16*)alloc((size_t)S_SEQ * HD_DIM * 2);
    __bf16* vtg  = (__bf16*)alloc((size_t)HD_DIM * S_SEQ * 2);   // V transposed [HD][S]
    __bf16* ob   = (__bf16*)alloc((size_t)S_SEQ * HD_DIM * 2);
    float*  x1   = (float*)alloc((size_t)S_SEQ * D_MODEL * 4);
    __bf16* x1b  = (__bf16*)alloc((size_t)S_SEQ * D_MODEL * 2);
    __bf16* tb   = (__bf16*)alloc((size_t)S_SEQ * F_FF * 2);
    __bf16* Wqt  = (__bf16*)alloc((size_t)HD_DIM * D_MODEL * 2); // [HD][D]
    __bf16* Wkt  = (__bf16*)alloc((size_t)HD_DIM * D_MODEL * 2);
    __bf16* Wvt  = (__bf16*)alloc((size_t)HD_DIM * D_MODEL * 2);
    __bf16* Wot  = (__bf16*)alloc((size_t)D_MODEL * HD_DIM * 2); // [D][HD]
    __bf16* W1t  = (__bf16*)alloc((size_t)F_FF * D_MODEL * 2);   // [F][D]
    __bf16* W2t  = (__bf16*)alloc((size_t)D_MODEL * F_FF * 2);   // [D][F]

    // 1) LayerNorm + RMSNorm
    ln_rms_kernel<<<S_SEQ, 256, 0, stream>>>(x, ln_g, ln_b, attn_g, xln, h);

    // 2) weight down-convert + transpose: W[rows][cols] f32 -> Wt[cols][rows] bf16
    cvt_transpose_bf16<<<dim3(HD_DIM / 32, D_MODEL / 32), 256, 0, stream>>>(Wq, Wqt, D_MODEL, HD_DIM);
    cvt_transpose_bf16<<<dim3(HD_DIM / 32, D_MODEL / 32), 256, 0, stream>>>(Wk, Wkt, D_MODEL, HD_DIM);
    cvt_transpose_bf16<<<dim3(HD_DIM / 32, D_MODEL / 32), 256, 0, stream>>>(Wv, Wvt, D_MODEL, HD_DIM);
    cvt_transpose_bf16<<<dim3(D_MODEL / 32, HD_DIM / 32), 256, 0, stream>>>(Wo, Wot, HD_DIM, D_MODEL);
    cvt_transpose_bf16<<<dim3(F_FF / 32, D_MODEL / 32), 256, 0, stream>>>(W1, W1t, D_MODEL, F_FF);
    cvt_transpose_bf16<<<dim3(D_MODEL / 32, F_FF / 32), 256, 0, stream>>>(W2, W2t, F_FF, D_MODEL);

    // 3) QKV projections (V written transposed [HD][S] for flash attention)
    {
        dim3 grid(HD_DIM / 128, S_SEQ / 128);
        gemm_bf16<false, false, false, false, true, false><<<grid, 256, 0, stream>>>(
            h, Wqt, nullptr, nullptr, nullptr, qb, nullptr, S_SEQ, HD_DIM, D_MODEL);
        gemm_bf16<false, false, false, false, true, false><<<grid, 256, 0, stream>>>(
            h, Wkt, nullptr, nullptr, nullptr, kbuf, nullptr, S_SEQ, HD_DIM, D_MODEL);
        gemm_bf16<false, false, false, false, false, true><<<grid, 256, 0, stream>>>(
            h, Wvt, nullptr, nullptr, nullptr, nullptr, vtg, S_SEQ, HD_DIM, D_MODEL);
    }

    // 4) causal flash attention
    flash_attn_kernel<<<dim3(S_SEQ / 64, N_HEADS), 128, 0, stream>>>(qb, kbuf, vtg, ob);

    // 5) output projection + residual (f32 x1 and bf16 x1b)
    {
        dim3 grid(D_MODEL / 128, S_SEQ / 128);
        gemm_bf16<false, false, true, true, true, false><<<grid, 256, 0, stream>>>(
            ob, Wot, nullptr, xln, x1, x1b, nullptr, S_SEQ, D_MODEL, HD_DIM);
    }

    // 6) FFN: t = gelu(x1 @ W1 + b1)
    {
        dim3 grid(F_FF / 128, S_SEQ / 128);
        gemm_bf16<true, true, false, false, true, false><<<grid, 256, 0, stream>>>(
            x1b, W1t, b1, nullptr, nullptr, tb, nullptr, S_SEQ, F_FF, D_MODEL);
    }

    // 7) FFN: out = t @ W2 + b2 + x1
    {
        dim3 grid(D_MODEL / 128, S_SEQ / 128);
        gemm_bf16<true, false, true, true, false, false><<<grid, 256, 0, stream>>>(
            tb, W2t, b2, x1, out, nullptr, nullptr, S_SEQ, D_MODEL, F_FF);
    }
}
